// RNNQNetwork_54872502174040
// MI455X (gfx1250) — matrix operand order repaired and checked by reading, not compiled, not run on hardware
//
#include <hip/hip_runtime.h>

// ---------------------------------------------------------------------------
// RNN Q-network for MI455X (gfx1250, wave32, WMMA 16x16x32 bf16 + TDM)
// ---------------------------------------------------------------------------
#define NB   64      // batch
#define LT   2048    // sequence length
#define INF  128     // input features
#define HD   256     // hidden
#define G3   768     // 3*HD
#define NA   18      // actions

typedef __bf16 bf16;
typedef __attribute__((ext_vector_type(16))) __bf16 v16bf;
typedef __attribute__((ext_vector_type(8)))  __bf16 v8bf;
typedef __attribute__((ext_vector_type(8)))  float  v8f;
typedef __attribute__((ext_vector_type(4)))  unsigned int ui32x4;
typedef __attribute__((ext_vector_type(8)))  int  i32x8;
typedef __attribute__((ext_vector_type(4)))  int  i32x4;

#if __has_builtin(__builtin_amdgcn_tensor_load_to_lds) && \
    __has_builtin(__builtin_amdgcn_s_wait_tensorcnt)
#define USE_TDM 1
#else
#define USE_TDM 0
#endif

// ---------------------------------------------------------------------------
// WMMA wrapper: D = A(16x32 bf16) * B(32x16 bf16) + C(16x16 f32)
// ---------------------------------------------------------------------------
__device__ __forceinline__ v8f wmma_bf16(v16bf a, v16bf b, v8f c) {
    return __builtin_amdgcn_wmma_f32_16x16x32_bf16(
        /*neg_a=*/false, a, /*neg_b=*/false, b,
        /*c_mod=*/(short)0, c, /*reuse_a=*/false, /*reuse_b=*/false);
}

// A-fragment (16x32, bf16) from a row-major bf16 matrix with leading dim ld.
// ISA layout: lanes 0-15 hold M=lane, K={kb..kb+7, kb+16..kb+23};
//             lanes 16-31 hold M=lane-16, K={kb+8..kb+15, kb+24..kb+31}.
__device__ __forceinline__ v16bf load_a_bf16(const bf16* base, int ld,
                                             int lane, int kb) {
    const int row = lane & 15;
    const int c0  = kb + ((lane & 16) ? 8 : 0);
    const bf16* p = base + (size_t)row * ld + c0;
    v8bf lo = *(const v8bf*)(p);
    v8bf hi = *(const v8bf*)(p + 16);
    v16bf r;
#pragma unroll
    for (int i = 0; i < 8; ++i) { r[i] = lo[i]; r[i + 8] = hi[i]; }
    return r;
}

// Same A-fragment but sourced from fp32 memory (convert on the fly).
__device__ __forceinline__ v16bf load_a_f32(const float* base, int ld,
                                            int lane, int kb) {
    const int row = lane & 15;
    const int c0  = kb + ((lane & 16) ? 8 : 0);
    const float* p = base + (size_t)row * ld + c0;
    v16bf r;
#pragma unroll
    for (int i = 0; i < 8; ++i) {
        r[i]     = (bf16)p[i];
        r[i + 8] = (bf16)p[i + 16];
    }
    return r;
}

// B-fragment (32x16, bf16) for out = X @ W^T with W row-major [n][k], ldk=ld.
// ISA layout: lane holds N-col = lane&15; K = {0..7, 16..23} (lanes 0-15) or
// {8..15, 24..31} (lanes 16-31) relative to kb -> two 16B loads.
__device__ __forceinline__ v16bf load_bT(const bf16* W, int ld,
                                         int n0, int lane, int kb) {
    const int wrow = n0 + (lane & 15);
    const int c    = kb + ((lane & 16) ? 8 : 0);
    const bf16* p = W + (size_t)wrow * ld + c;
    v8bf lo = *(const v8bf*)(p);
    v8bf hi = *(const v8bf*)(p + 16);
    v16bf r;
#pragma unroll
    for (int i = 0; i < 8; ++i) { r[i] = lo[i]; r[i + 8] = hi[i]; }
    return r;
}

__device__ __forceinline__ float sigmoidf_(float x) {
    return 1.0f / (1.0f + __expf(-x));
}

#if USE_TDM
// LDS byte offset = low 32 bits of the flat shared-aperture address
// (ISA 10.2).
__device__ __forceinline__ unsigned lds_off(const void* p) {
    return (unsigned)(unsigned long long)p;
}

// Issue one 2D TDM tile load (bf16 elements): rows x dim0, row stride
// stride_elems (in elements), full-width tile, into LDS at lds_byte_off.
// Descriptor per CDNA5 ISA ch.8 (D# group0/group1; groups 2-3 unused/zero).
__device__ __forceinline__ void tdm_load_2d(const void* gptr,
                                            unsigned lds_byte_off,
                                            unsigned dim0, unsigned rows,
                                            unsigned long long stride_elems) {
    unsigned long long ga = (unsigned long long)gptr;
    ui32x4 g0 = { 1u,                                   // count=1, user mode
                  lds_byte_off,                         // lds_addr
                  (unsigned)(ga & 0xFFFFFFFFu),         // global_addr[31:0]
                  (unsigned)(((ga >> 32) & 0x01FFFFFFu) // global_addr[56:32]
                             | 0x80000000u) };          // type=2 ("image")
    i32x8 g1;
    g1[0] = (int)(1u << 16);                            // data_size=1 -> 2B
    g1[1] = (int)((dim0 & 0xFFFFu) << 16);              // tensor_dim0[15:0]
    g1[2] = (int)((dim0 >> 16) | ((rows & 0xFFFFu) << 16));  // d0 hi | d1 lo
    g1[3] = (int)((rows >> 16) | ((dim0 & 0xFFFFu) << 16));  // d1 hi | tile_dim0
    g1[4] = (int)(rows & 0xFFFFu);                      // tile_dim1 | tile_dim2=0
    g1[5] = (int)(stride_elems & 0xFFFFFFFFu);          // dim0_stride lo
    g1[6] = (int)((stride_elems >> 32) & 0xFFFFu);      // dim0_stride hi
    g1[7] = 0;
    i32x4 z4 = {0, 0, 0, 0};
#if __clang_major__ >= 23
    i32x8 z8 = {0, 0, 0, 0, 0, 0, 0, 0};
    __builtin_amdgcn_tensor_load_to_lds(g0, g1, z4, z4, z8, 0);
#else
    __builtin_amdgcn_tensor_load_to_lds(g0, g1, z4, z4, 0);
#endif
}
#endif  // USE_TDM

// ---------------------------------------------------------------------------
// Kernel 0: fp32 -> bf16 weight conversion (w_ih, w_hh, fc0, fc1, fc2 padded)
// ---------------------------------------------------------------------------
__global__ void cvt_kernel(const float* __restrict__ w_ih,
                           const float* __restrict__ w_hh,
                           const float* __restrict__ fc0,
                           const float* __restrict__ fc1,
                           const float* __restrict__ fc2,
                           bf16* __restrict__ w_ih_b, bf16* __restrict__ w_hh_b,
                           bf16* __restrict__ fc0_b,  bf16* __restrict__ fc1_b,
                           bf16* __restrict__ fc2_b) {
    const int i = blockIdx.x * blockDim.x + threadIdx.x;
    if (i < G3 * INF) w_ih_b[i] = (bf16)w_ih[i];
    if (i < G3 * HD)  w_hh_b[i] = (bf16)w_hh[i];
    if (i < HD * HD) { fc0_b[i] = (bf16)fc0[i]; fc1_b[i] = (bf16)fc1[i]; }
    if (i < 32 * HD)  fc2_b[i] = (bf16)((i < NA * HD) ? fc2[i] : 0.0f);
}

// ---------------------------------------------------------------------------
// Kernel 1: gi = state @ w_ih^T + b_ih   (M=131072, K=128, N=768), bf16 out.
// ---------------------------------------------------------------------------
__global__ __launch_bounds__(128) void proj_kernel(
    const float* __restrict__ state, const bf16* __restrict__ w_ih_b,
    const float* __restrict__ b_ih, bf16* __restrict__ gi_b) {
    const int lane  = threadIdx.x & 31;
    const int wv    = threadIdx.x >> 5;
    const int strip = blockIdx.x * 4 + wv;      // 0..8191 (16-row strips)
    const int n0    = blockIdx.y * 64;          // 0,64,...,704
    const float* Abase = state + (size_t)strip * 16 * INF;

    v8f acc[4] = {{}, {}, {}, {}};
#pragma unroll
    for (int kt = 0; kt < 4; ++kt) {
        v16bf a = load_a_f32(Abase, INF, lane, kt * 32);
#pragma unroll
        for (int nt = 0; nt < 4; ++nt) {
            v16bf b = load_bT(w_ih_b, INF, n0 + nt * 16, lane, kt * 32);
            acc[nt] = wmma_bf16(a, b, acc[nt]);
        }
    }
#pragma unroll
    for (int nt = 0; nt < 4; ++nt) {
        const int col  = n0 + nt * 16 + (lane & 15);
        const float bv = b_ih[col];
#pragma unroll
        for (int v = 0; v < 8; ++v) {
            const size_t mrow = (size_t)strip * 16 + v + ((lane & 16) ? 8 : 0);
            gi_b[mrow * G3 + col] = (bf16)(acc[nt][v] + bv);
        }
    }
}

// ---------------------------------------------------------------------------
// Kernel 2: resetting-GRU recurrence + padded-tail replay.
// grid = 4 blocks x 512 threads (16 waves). Block owns 16 batch rows.
// Wave w owns hidden columns [16w,16w+16) of ALL three gates; w_hh
// B-fragments live in registers. gi[t] tiles are double-buffered into LDS
// by the Tensor Data Mover, overlapped with step t-1's compute.
// ---------------------------------------------------------------------------
__global__ __launch_bounds__(512) void gru_kernel(
    const float* __restrict__ hs, const int* __restrict__ dones,
    const float* __restrict__ b_ih, const float* __restrict__ b_hh,
    const bf16* __restrict__ w_hh_b, const bf16* __restrict__ gi_b,
    float* __restrict__ gru_out, float* __restrict__ next_hx) {
    __shared__ float h_f[16 * HD];
    __shared__ bf16  h_b[16 * HD];
    __shared__ int   c_s[16];
#if USE_TDM
    __shared__ bf16  gi_s[2 * 16 * G3];     // 48 KB double buffer
#endif

    const int tid  = threadIdx.x;
    const int lane = tid & 31;
    const int wv   = tid >> 5;       // 0..15
    const int nb0  = blockIdx.x * 16;
    const int jcol = wv * 16 + (lane & 15);   // this wave's hidden column

    // --- init hidden state ---------------------------------------------------
    for (int e = tid; e < 16 * HD; e += 512) {
        const int m = e >> 8, j = e & 255;
        const float v = hs[(size_t)(nb0 + m) * HD + j];
        h_f[e] = v; h_b[e] = (bf16)v;
    }
    if (tid < 16) c_s[tid] = 0;
    __syncthreads();

    // --- last-done index per row (for the padded tail) -----------------------
    for (int m = 0; m < 16; ++m)
        for (int l = tid; l < LT; l += 512)
            if (dones[(size_t)(nb0 + m) * LT + l]) atomicMax(&c_s[m], l);

    // --- preload w_hh B-fragments into registers (3 gates x 8 k-tiles) -------
    v16bf Br[8], Bz[8], Bn[8];
#pragma unroll
    for (int kt = 0; kt < 8; ++kt) {
        Br[kt] = load_bT(w_hh_b, HD, 0 * HD + wv * 16, lane, kt * 32);
        Bz[kt] = load_bT(w_hh_b, HD, 1 * HD + wv * 16, lane, kt * 32);
        Bn[kt] = load_bT(w_hh_b, HD, 2 * HD + wv * 16, lane, kt * 32);
    }
    const float bhr = b_hh[jcol], bhz = b_hh[HD + jcol], bhn = b_hh[2 * HD + jcol];
    const float bir = b_ih[jcol], biz = b_ih[HD + jcol], bin_ = b_ih[2 * HD + jcol];

#if USE_TDM
    const unsigned gi_s_off = lds_off(gi_s);
    if (wv == 0)   // prologue: DMA gi tile for t = 0 into buffer 0
        tdm_load_2d(gi_b + (size_t)nb0 * LT * G3, gi_s_off, G3, 16,
                    (unsigned long long)LT * G3);
#endif

    // --- main scan over time -------------------------------------------------
    for (int t = 0; t < LT; ++t) {
        __syncthreads();   // previous step's h stores + gi-buffer reads done
#if USE_TDM
        if (wv == 0 && t + 1 < LT)   // overlap: DMA gi[t+1] into other buffer
            tdm_load_2d(gi_b + ((size_t)nb0 * LT + (t + 1)) * G3,
                        gi_s_off + (unsigned)(((t + 1) & 1) * (16 * G3 * 2)),
                        G3, 16, (unsigned long long)LT * G3);
#endif
        {   // episode reset before the step
            const int m = tid >> 5, j0 = lane * 8;
            if (dones[(size_t)(nb0 + m) * LT + t]) {
#pragma unroll
                for (int i = 0; i < 8; ++i) {
                    h_f[m * HD + j0 + i] = 0.0f;
                    h_b[m * HD + j0 + i] = (bf16)0.0f;
                }
            }
#if !USE_TDM
            if (t + 1 < LT)
                __builtin_prefetch(gi_b + ((size_t)(nb0 + m) * LT + t + 1) * G3
                                        + (size_t)lane * 24, 0, 1);
#endif
        }
#if USE_TDM
        if (wv == 0) {   // TENSORcnt is in-order: <=1 means tile t has landed
            if (t + 1 < LT) __builtin_amdgcn_s_wait_tensorcnt(1);
            else            __builtin_amdgcn_s_wait_tensorcnt(0);
        }
#endif
        __syncthreads();

        // read A = h (16x256 bf16) once, before anyone rewrites it
        v16bf At[8];
#pragma unroll
        for (int kt = 0; kt < 8; ++kt) At[kt] = load_a_bf16(h_b, HD, lane, kt * 32);
        __syncthreads();

        v8f ar = {}, az = {}, an = {};
#pragma unroll
        for (int kt = 0; kt < 8; ++kt) {
            ar = wmma_bf16(At[kt], Br[kt], ar);
            az = wmma_bf16(At[kt], Bz[kt], az);
            an = wmma_bf16(At[kt], Bn[kt], an);
        }

        // fused gate math directly on WMMA accumulators
#pragma unroll
        for (int v = 0; v < 8; ++v) {
            const int m = v + ((lane & 16) ? 8 : 0);
            const size_t grow = (size_t)(nb0 + m) * LT + t;
#if USE_TDM
            const bf16* gp = gi_s + (size_t)(t & 1) * (16 * G3) + m * G3;
#else
            const bf16* gp = gi_b + grow * G3;
#endif
            const float ir  = (float)gp[jcol];
            const float iz  = (float)gp[HD + jcol];
            const float in_ = (float)gp[2 * HD + jcol];
            const float hp  = h_f[m * HD + jcol];
            const float r   = sigmoidf_(ir + ar[v] + bhr);
            const float z   = sigmoidf_(iz + az[v] + bhz);
            const float nn  = tanhf(in_ + r * (an[v] + bhn));
            const float hn  = (1.0f - z) * nn + z * hp;
            h_f[m * HD + jcol] = hn;
            h_b[m * HD + jcol] = (bf16)hn;
            gru_out[grow * HD + jcol] = hn;
        }
    }

    // --- padded-batch tail: c extra zero-input steps, predicated per row -----
    __syncthreads();
    int cmax = 0;
#pragma unroll
    for (int m = 0; m < 16; ++m) cmax = max(cmax, c_s[m]);

    for (int k = 0; k < cmax; ++k) {
        __syncthreads();
        v16bf At[8];
#pragma unroll
        for (int kt = 0; kt < 8; ++kt) At[kt] = load_a_bf16(h_b, HD, lane, kt * 32);
        __syncthreads();

        v8f ar = {}, az = {}, an = {};
#pragma unroll
        for (int kt = 0; kt < 8; ++kt) {
            ar = wmma_bf16(At[kt], Br[kt], ar);
            az = wmma_bf16(At[kt], Bz[kt], az);
            an = wmma_bf16(At[kt], Bn[kt], an);
        }
#pragma unroll
        for (int v = 0; v < 8; ++v) {
            const int m = v + ((lane & 16) ? 8 : 0);
            const float hp = h_f[m * HD + jcol];
            const float r  = sigmoidf_(bir + ar[v] + bhr);
            const float z  = sigmoidf_(biz + az[v] + bhz);
            const float nn = tanhf(bin_ + r * (an[v] + bhn));
            const float hn = (1.0f - z) * nn + z * hp;
            if (k < c_s[m]) {
                h_f[m * HD + jcol] = hn;
                h_b[m * HD + jcol] = (bf16)hn;
            }
        }
    }
    __syncthreads();
    for (int e = tid; e < 16 * HD; e += 512) {
        const int m = e >> 8, j = e & 255;
        next_hx[(size_t)(nb0 + m) * HD + j] = h_f[e];
    }
}

// ---------------------------------------------------------------------------
// Kernel 3: fused FC head. All weights LDS-resident (304 KB of the 320 KB
// WGP LDS), DMA'd in by the Tensor Data Mover. Persistent blocks stream
// 32-row activation tiles.
// ---------------------------------------------------------------------------
__device__ __forceinline__ void fc_stage(const bf16* Xin, const bf16* W,
                                         const float* __restrict__ bias,
                                         bf16* Yout, int lane, int wv) {
#pragma unroll
    for (int nn = 0; nn < 2; ++nn) {
        const int n0 = (wv + nn * 8) * 16;
        const float bv = bias[n0 + (lane & 15)];
#pragma unroll
        for (int mt = 0; mt < 2; ++mt) {
            v8f acc = {};
#pragma unroll
            for (int kt = 0; kt < 8; ++kt) {
                v16bf a = load_a_bf16(Xin + mt * 16 * HD, HD, lane, kt * 32);
                v16bf b = load_bT(W, HD, n0, lane, kt * 32);
                acc = wmma_bf16(a, b, acc);
            }
            const int col = n0 + (lane & 15);
#pragma unroll
            for (int v = 0; v < 8; ++v) {
                const int m = mt * 16 + v + ((lane & 16) ? 8 : 0);
                Yout[m * HD + col] = (bf16)fmaxf(acc[v] + bv, 0.0f);
            }
        }
    }
}

__global__ __launch_bounds__(256) void fc_kernel(
    const bf16* __restrict__ fc0_b16, const bf16* __restrict__ fc1_b16,
    const bf16* __restrict__ fc2_b16,
    const float* __restrict__ fc0_bias, const float* __restrict__ fc1_bias,
    const float* __restrict__ fc2_bias,
    const float* __restrict__ gru_out, float* __restrict__ q_out) {
    extern __shared__ char smem[];
    bf16* W0 = (bf16*)smem;            // 256x256
    bf16* W1 = W0 + HD * HD;           // 256x256
    bf16* W2 = W1 + HD * HD;           // 32x256 (padded fc2)
    bf16* X  = W2 + 32 * HD;           // 32x256 activation tile
    bf16* Y  = X + 32 * HD;            // 32x256 activation tile

    const int tid = threadIdx.x, lane = tid & 31, wv = tid >> 5;

#if USE_TDM
    if (wv == 0) {   // three TDM descriptors; TENSORcnt is in-order per wave
        tdm_load_2d(fc0_b16, lds_off(W0), HD, HD, HD);
        tdm_load_2d(fc1_b16, lds_off(W1), HD, HD, HD);
        tdm_load_2d(fc2_b16, lds_off(W2), HD, 32, HD);
        __builtin_amdgcn_s_wait_tensorcnt(0);
    }
#else
    for (int e = tid; e < (HD * HD) / 8; e += 256)
        ((v8bf*)W0)[e] = ((const v8bf*)fc0_b16)[e];
    for (int e = tid; e < (HD * HD) / 8; e += 256)
        ((v8bf*)W1)[e] = ((const v8bf*)fc1_b16)[e];
    for (int e = tid; e < (32 * HD) / 8; e += 256)
        ((v8bf*)W2)[e] = ((const v8bf*)fc2_b16)[e];
#endif

    const int MT = (NB * LT) / 32;   // 4096 row-tiles
    for (int tIdx = blockIdx.x; tIdx < MT; tIdx += gridDim.x) {
        const size_t r0 = (size_t)tIdx * 32;
        __syncthreads();   // weights ready / previous tile's reads done
        for (int e = tid; e < 32 * HD; e += 256)
            X[e] = (bf16)fmaxf(gru_out[r0 * HD + e], 0.0f);   // relu(gru_out)
        __syncthreads();

        fc_stage(X, W0, fc0_bias, Y, lane, wv);   // x1 in Y
        __syncthreads();
        fc_stage(Y, W1, fc1_bias, X, lane, wv);   // x2 in X
        __syncthreads();

        if (wv < 4) {   // q = x2 @ fc2^T + b2, cols 0..17 valid
            const int nt = wv & 1, mt = wv >> 1;
            v8f acc = {};
#pragma unroll
            for (int kt = 0; kt < 8; ++kt) {
                v16bf a = load_a_bf16(X + mt * 16 * HD, HD, lane, kt * 32);
                v16bf b = load_bT(W2, HD, nt * 16, lane, kt * 32);
                acc = wmma_bf16(a, b, acc);
            }
            const int col = nt * 16 + (lane & 15);
            if (col < NA) {
                const float bv = fc2_bias[col];
#pragma unroll
                for (int v = 0; v < 8; ++v) {
                    const int m = mt * 16 + v + ((lane & 16) ? 8 : 0);
                    q_out[(r0 + m) * NA + col] = acc[v] + bv;
                }
            }
        }
    }
}

// ---------------------------------------------------------------------------
// Host-side launcher
// ---------------------------------------------------------------------------
extern "C" void kernel_launch(void* const* d_in, const int* in_sizes, int n_in,
                              void* d_out, int out_size, void* d_ws, size_t ws_size,
                              hipStream_t stream) {
    const float* state = (const float*)d_in[0];
    const float* hs    = (const float*)d_in[1];
    const int*   dones = (const int*)d_in[2];
    const float* w_ih  = (const float*)d_in[3];
    const float* w_hh  = (const float*)d_in[4];
    const float* b_ih  = (const float*)d_in[5];
    const float* b_hh  = (const float*)d_in[6];
    const float* fc0_w = (const float*)d_in[7];
    const float* fc0_b = (const float*)d_in[8];
    const float* fc1_w = (const float*)d_in[9];
    const float* fc1_b = (const float*)d_in[10];
    const float* fc2_w = (const float*)d_in[11];
    const float* fc2_b = (const float*)d_in[12];

    float* q_out    = (float*)d_out;                           // [N,L,A]
    float* gru_out  = q_out + (size_t)NB * LT * NA;            // [N,L,H]
    float* next_hx  = gru_out + (size_t)NB * LT * HD;          // [1,N,H]

    // workspace layout (bytes)
    char* ws = (char*)d_ws;
    bf16* w_ih_b = (bf16*)(ws + 0);            // 768*128*2  = 196608
    bf16* w_hh_b = (bf16*)(ws + 196608);       // 768*256*2  = 393216
    bf16* fc0_bw = (bf16*)(ws + 589824);       // 256*256*2  = 131072
    bf16* fc1_bw = (bf16*)(ws + 720896);       // 256*256*2  = 131072
    bf16* fc2_bw = (bf16*)(ws + 851968);       //  32*256*2  =  16384
    bf16* gi_b   = (bf16*)(ws + 1048576);      // 131072*768*2 = 192 MiB

    hipLaunchKernelGGL(cvt_kernel, dim3((G3 * HD + 255) / 256), dim3(256), 0, stream,
                       w_ih, w_hh, fc0_w, fc1_w, fc2_w,
                       w_ih_b, w_hh_b, fc0_bw, fc1_bw, fc2_bw);

    hipLaunchKernelGGL(proj_kernel, dim3((NB * LT / 16) / 4, G3 / 64), dim3(128),
                       0, stream, state, w_ih_b, b_ih, gi_b);

    hipLaunchKernelGGL(gru_kernel, dim3(NB / 16), dim3(512), 0, stream,
                       hs, dones, b_ih, b_hh, w_hh_b, gi_b, gru_out, next_hx);

    const size_t smem_fc = (size_t)(HD * HD + HD * HD + 32 * HD + 32 * HD + 32 * HD)
                           * sizeof(bf16);   // 304 KiB
    hipLaunchKernelGGL(fc_kernel, dim3(512), dim3(256), smem_fc, stream,
                       fc0_bw, fc1_bw, fc2_bw, fc0_b, fc1_b, fc2_b,
                       gru_out, q_out);
}
